// MatrixDiTBlock_39256001085955
// MI455X (gfx1250) — matrix operand order repaired and checked by reading, not compiled, not run on hardware
//
#include <hip/hip_runtime.h>

typedef __attribute__((ext_vector_type(16))) __bf16 v16bf;
typedef __attribute__((ext_vector_type(8)))  __bf16 v8bf;
typedef __attribute__((ext_vector_type(8)))  float  v8f;

// async-copy pointee type per clang builtin prototype: int __vector_size__(16)
typedef int v4i_g __attribute__((vector_size(16)));
typedef __attribute__((address_space(1))) v4i_g as1_v4i;
typedef __attribute__((address_space(3))) v4i_g as3_v4i;

#if defined(__gfx1250__) && __has_builtin(__builtin_amdgcn_global_load_async_to_lds_b128)
#define HAVE_ASYNC_LDS 1
#else
#define HAVE_ASYNC_LDS 0
#endif

#if HAVE_ASYNC_LDS
#if __has_builtin(__builtin_amdgcn_s_wait_asynccnt)
#define WAIT_ASYNC(n) __builtin_amdgcn_s_wait_asynccnt(n)
#else
#define WAIT_ASYNC(n) asm volatile("s_wait_asynccnt %0" ::"n"(n) : "memory")
#endif
#else
#define WAIT_ASYNC(n)
#endif

#define SHFL16(lo, hi) __builtin_shufflevector(lo, hi, 0,1,2,3,4,5,6,7,8,9,10,11,12,13,14,15)

__device__ __forceinline__ float gelu_tanh(float x) {
    float x3 = x * x * x;
    return 0.5f * x * (1.0f + tanhf(0.7978845608028654f * (x + 0.044715f * x3)));
}

// ---------------------------------------------------------------------------
// Elementwise helpers
// ---------------------------------------------------------------------------
__global__ void silu_bf16_kernel(const float* __restrict__ in, __bf16* __restrict__ out, long long n) {
    long long i = (long long)blockIdx.x * blockDim.x + threadIdx.x;
    long long st = (long long)gridDim.x * blockDim.x;
    for (; i < n; i += st) {
        float v = in[i];
        out[i] = (__bf16)(v / (1.0f + __expf(-v)));
    }
}

__global__ void cvt_bf16_kernel(const float* __restrict__ in, __bf16* __restrict__ out, long long n) {
    long long i = (long long)blockIdx.x * blockDim.x + threadIdx.x;
    long long st = (long long)gridDim.x * blockDim.x;
    for (; i < n; i += st) out[i] = (__bf16)in[i];
}

// in: (R, Cc) row-major f32  ->  out: (Cc, R) row-major bf16  (out[c][r] = in[r][c])
__global__ void tconv_bf16_kernel(const float* __restrict__ in, __bf16* __restrict__ out, int R, int Cc) {
    long long n = (long long)R * Cc;
    long long i = (long long)blockIdx.x * blockDim.x + threadIdx.x;
    long long st = (long long)gridDim.x * blockDim.x;
    for (; i < n; i += st) {
        int r = (int)(i / Cc);
        int c = (int)(i % Cc);
        out[(long long)c * R + r] = (__bf16)in[i];
    }
}

// out = base + m[row, 2048+c] * v    (gated residual; m row stride 3072, gate is 3rd chunk)
__global__ void residual_gate_kernel(const float* __restrict__ base, const float* __restrict__ m,
                                     const float* __restrict__ v, float* __restrict__ out, long long n) {
    long long i = (long long)blockIdx.x * blockDim.x + threadIdx.x;
    long long st = (long long)gridDim.x * blockDim.x;
    for (; i < n; i += st) {
        long long row = i >> 10;
        int cc = (int)(i & 1023);
        out[i] = base[i] + m[row * 3072 + 2048 + cc] * v[i];
    }
}

// ---------------------------------------------------------------------------
// adaLN: y = LN(x) * (1 + scale) + shift ; scale=m[row,1024+c], shift=m[row,c]
// ---------------------------------------------------------------------------
__global__ __launch_bounds__(256)
void adaln_kernel(const float* __restrict__ x, const float* __restrict__ m,
                  float* __restrict__ yf, __bf16* __restrict__ yh) {
    __shared__ float red[256];
    const int row = blockIdx.x;
    const int t = threadIdx.x;
    const float4 xv = ((const float4*)(x + (long long)row * 1024))[t];
    red[t] = xv.x + xv.y + xv.z + xv.w;
    __syncthreads();
    for (int o = 128; o > 0; o >>= 1) { if (t < o) red[t] += red[t + o]; __syncthreads(); }
    const float mean = red[0] * (1.0f / 1024.0f);
    __syncthreads();
    float d0 = xv.x - mean, d1 = xv.y - mean, d2 = xv.z - mean, d3 = xv.w - mean;
    red[t] = d0 * d0 + d1 * d1 + d2 * d2 + d3 * d3;
    __syncthreads();
    for (int o = 128; o > 0; o >>= 1) { if (t < o) red[t] += red[t + o]; __syncthreads(); }
    const float rstd = rsqrtf(red[0] * (1.0f / 1024.0f) + 1e-6f);
    const float* mr = m + (long long)row * 3072;
    float dv[4] = {d0, d1, d2, d3};
#pragma unroll
    for (int k = 0; k < 4; k++) {
        int cc = t * 4 + k;
        float y = dv[k] * rstd * (1.0f + mr[1024 + cc]) + mr[cc];
        yf[(long long)row * 1024 + cc] = y;
        yh[(long long)row * 1024 + cc] = (__bf16)y;
    }
}

// ---------------------------------------------------------------------------
// Generic bf16 GEMM: C[bz] = A[bz] (MxK) @ B[bz] (KxN) (+bias) (+gelu)
// 128x256x32 block tiles, 256 threads = 8 waves (2x4), wave does 64x64 = 4x4
// WMMA tiles. Double-buffered LDS fed by async global->LDS copies (ASYNCcnt).
// ---------------------------------------------------------------------------
#define BM 128
#define BN 256
#define BK 32
#define LDA_S 40    // As row stride (halves), 80B = 5x16B
#define LDB_S 264   // Bs row stride (halves), 528B = 33x16B

template<bool OUT_BF16, bool HAS_BIAS, bool DO_GELU>
__global__ __launch_bounds__(256)
void gemm_bf16_kernel(const __bf16* __restrict__ A, const __bf16* __restrict__ B,
                      const float* __restrict__ bias, void* __restrict__ Cout,
                      int M, int N, int K,
                      long long sA, long long sB, long long sC) {
    __shared__ __bf16 As[2][BM * LDA_S];
    __shared__ __bf16 Bs[2][BK * LDB_S];

    const int tid = threadIdx.x;
    const int lane = tid & 31;
    const int wave = tid >> 5;
    const int wm = wave >> 2;   // 0..1 (row 64-block)
    const int wn = wave & 3;    // 0..3 (col 64-block)

    const int bn = blockIdx.x * BN;
    const int bm = blockIdx.y * BM;
    const long long bz = blockIdx.z;

    const __bf16* Ab = A + bz * sA;
    const __bf16* Bb = B + bz * sB;

    v8f acc[4][4];
#pragma unroll
    for (int i = 0; i < 4; i++)
#pragma unroll
        for (int j = 0; j < 4; j++) acc[i][j] = (v8f){0.f, 0.f, 0.f, 0.f, 0.f, 0.f, 0.f, 0.f};

    const int ar = tid >> 1;          // 0..127 (A tile row)
    const int ac = (tid & 1) * 16;    // 0 / 16
    const int br = tid >> 3;          // 0..31  (B tile k-row)
    const int bc = (tid & 7) * 32;    // 0..224

    const int lr = lane >> 4;         // 0/1
    const int lc = lane & 15;
    const int akb = lr * 8;           // A frag K base per documented layout

    // stage tile (k0) into buffer buf: 6 x 16B per thread
    auto stage = [&](int k0, int buf) {
        const __bf16* sa = Ab + (long long)(bm + ar) * K + (k0 + ac);
        __bf16* da = &As[buf][ar * LDA_S + ac];
        const __bf16* sb = Bb + (long long)(k0 + br) * N + (bn + bc);
        __bf16* db = &Bs[buf][br * LDB_S + bc];
#if HAVE_ASYNC_LDS
        __builtin_amdgcn_global_load_async_to_lds_b128((as1_v4i*)sa, (as3_v4i*)da, 0, 0);
        __builtin_amdgcn_global_load_async_to_lds_b128((as1_v4i*)(sa + 8), (as3_v4i*)(da + 8), 0, 0);
        __builtin_amdgcn_global_load_async_to_lds_b128((as1_v4i*)sb, (as3_v4i*)db, 0, 0);
        __builtin_amdgcn_global_load_async_to_lds_b128((as1_v4i*)(sb + 8), (as3_v4i*)(db + 8), 0, 0);
        __builtin_amdgcn_global_load_async_to_lds_b128((as1_v4i*)(sb + 16), (as3_v4i*)(db + 16), 0, 0);
        __builtin_amdgcn_global_load_async_to_lds_b128((as1_v4i*)(sb + 24), (as3_v4i*)(db + 24), 0, 0);
#else
        uint4 a0 = *(const uint4*)(sa);
        uint4 a1 = *(const uint4*)(sa + 8);
        uint4 b0 = *(const uint4*)(sb);
        uint4 b1 = *(const uint4*)(sb + 8);
        uint4 b2 = *(const uint4*)(sb + 16);
        uint4 b3 = *(const uint4*)(sb + 24);
        *(uint4*)(da) = a0;
        *(uint4*)(da + 8) = a1;
        *(uint4*)(db) = b0;
        *(uint4*)(db + 8) = b1;
        *(uint4*)(db + 16) = b2;
        *(uint4*)(db + 24) = b3;
#endif
    };

    const int nk = K / BK;
    stage(0, 0);
    for (int t = 0; t < nk; t++) {
        const int buf = t & 1;
        if (t + 1 < nk) {
            stage((t + 1) * BK, buf ^ 1);
            WAIT_ASYNC(6);   // previous tile's 6 copies done; prefetch still in flight
        } else {
            WAIT_ASYNC(0);
        }
        __syncthreads();

        v16bf afr[4];
#pragma unroll
        for (int mi = 0; mi < 4; mi++) {
            int r = wm * 64 + mi * 16 + lc;   // M row
            v8bf lo = *(const v8bf*)(&As[buf][r * LDA_S + akb]);
            v8bf hi = *(const v8bf*)(&As[buf][r * LDA_S + akb + 16]);
            afr[mi] = SHFL16(lo, hi);
        }
        v16bf bfr[4];
#pragma unroll
        for (int ni = 0; ni < 4; ni++) {
            int ccol = wn * 64 + ni * 16;     // N col base; lane%16 = N col, elements = K (table order)
#pragma unroll
            for (int s = 0; s < 16; s++)
                bfr[ni][s] = Bs[buf][(akb + s + (s >= 8 ? 8 : 0)) * LDB_S + ccol + lc];
        }
#pragma unroll
        for (int mi = 0; mi < 4; mi++)
#pragma unroll
            for (int ni = 0; ni < 4; ni++)
                acc[mi][ni] = __builtin_amdgcn_wmma_f32_16x16x32_bf16(
                    false, afr[mi], false, bfr[ni], (short)0, acc[mi][ni], false, false);
        __syncthreads();
    }

    // epilogue: VGPR i -> M = i + 8*lr, N = lc (documented C layout)
#pragma unroll
    for (int mi = 0; mi < 4; mi++)
#pragma unroll
        for (int ni = 0; ni < 4; ni++) {
            int rbase = bm + wm * 64 + mi * 16 + lr * 8;
            int col = bn + wn * 64 + ni * 16 + lc;
            float bv = HAS_BIAS ? bias[col] : 0.0f;
#pragma unroll
            for (int i = 0; i < 8; i++) {
                long long idx = bz * sC + (long long)(rbase + i) * N + col;
                float v = acc[mi][ni][i] + bv;
                if (DO_GELU) v = gelu_tanh(v);
                if (OUT_BF16) ((__bf16*)Cout)[idx] = (__bf16)v;
                else          ((float*)Cout)[idx] = v;
            }
        }
}

// ---------------------------------------------------------------------------
// Attention per head (B*MCH*NRH = 32 heads), L=16, hw = HC*HR = 64*256 = 16384
// qkv layout: (B, L, 786432) flat, q/k/v offsets s*262144 + mch*65536 + hc*1024 + nrh*256 + hr
// writes o (B, L, 256, 1024) bf16: o[b][l][mch*64+hc][nrh*256+hr]
// ---------------------------------------------------------------------------
__global__ __launch_bounds__(256)
void attention_kernel(const __bf16* __restrict__ qkv, __bf16* __restrict__ o) {
    __shared__ float qs[16][260];
    __shared__ float ks[16][260];
    __shared__ float attn_s[16][16];

    const int hd = blockIdx.x;
    const int b = hd >> 4;
    const int mch = (hd >> 2) & 3;
    const int nrh = hd & 3;
    const int t = threadIdx.x;
    const int i = t >> 4, j = t & 15;

    const long long bbase = (long long)b * 16 * 786432 + (long long)mch * 65536 + (long long)nrh * 256;

    float acc = 0.0f;
    for (int hw0 = 0; hw0 < 16384; hw0 += 256) {
        const int hc = hw0 >> 8;
        const long long off = bbase + (long long)hc * 1024 + t;  // hr = t
        for (int r = 0; r < 16; r++) {
            long long lo = off + (long long)r * 786432;
            qs[r][t] = (float)qkv[lo];
            ks[r][t] = (float)qkv[lo + 262144];
        }
        __syncthreads();
        for (int e = 0; e < 256; e++) acc += qs[i][e] * ks[j][e];
        __syncthreads();
    }
    attn_s[i][j] = acc * 128.0f;  // divide by SCALE = (HC*HR)^-0.5 = 1/128
    __syncthreads();

    if (t < 16) {
        float mx = -1e30f;
        for (int jj = 0; jj < 16; jj++) mx = fmaxf(mx, attn_s[t][jj]);
        float s = 0.0f, e[16];
        for (int jj = 0; jj < 16; jj++) { e[jj] = __expf(attn_s[t][jj] - mx); s += e[jj]; }
        float inv = 1.0f / s;
        for (int jj = 0; jj < 16; jj++) attn_s[t][jj] = e[jj] * inv;
    }
    __syncthreads();

    for (int hw0 = 0; hw0 < 16384; hw0 += 256) {
        const int hc = hw0 >> 8;
        const long long voff = bbase + 2LL * 262144 + (long long)hc * 1024 + t;
        float ov[16];
#pragma unroll
        for (int ii = 0; ii < 16; ii++) ov[ii] = 0.0f;
        for (int jj = 0; jj < 16; jj++) {
            float vj = (float)qkv[voff + (long long)jj * 786432];
#pragma unroll
            for (int ii = 0; ii < 16; ii++) ov[ii] += attn_s[ii][jj] * vj;
        }
        for (int ii = 0; ii < 16; ii++) {
            long long oidx = (((long long)(b * 16 + ii) * 256) + mch * 64 + hc) * 1024 + nrh * 256 + t;
            o[oidx] = (__bf16)ov[ii];
        }
    }
}

// ---------------------------------------------------------------------------
// Host launcher
// ---------------------------------------------------------------------------
extern "C" void kernel_launch(void* const* d_in, const int* in_sizes, int n_in,
                              void* d_out, int out_size, void* d_ws, size_t ws_size,
                              hipStream_t stream) {
    (void)in_sizes; (void)n_in; (void)out_size; (void)ws_size;
    const float* x      = (const float*)d_in[0];
    const float* c      = (const float*)d_in[1];
    const float* mod1_w = (const float*)d_in[2];
    const float* mod1_b = (const float*)d_in[3];
    const float* qkv_u  = (const float*)d_in[4];
    const float* qkv_v  = (const float*)d_in[5];
    const float* proj_u = (const float*)d_in[6];
    const float* proj_v = (const float*)d_in[7];
    const float* mod2_w = (const float*)d_in[8];
    const float* mod2_b = (const float*)d_in[9];
    const float* fc1_w  = (const float*)d_in[10];
    const float* fc1_b  = (const float*)d_in[11];
    const float* fc2_w  = (const float*)d_in[12];
    const float* fc2_b  = (const float*)d_in[13];
    float* out = (float*)d_out;

    const long long BNC = 8388608;         // 2*4096*1024
    const long long M3C = 25165824;        // 8192*3072
    const long long QKV = 25165824;        // 32*256*3072
    const long long HEL = 33554432;        // 8192*4096

    char* ws = (char*)d_ws;
    size_t off = 0;
    auto alloc = [&](size_t bytes) -> void* {
        void* p = ws + off;
        off = (off + bytes + 255) & ~(size_t)255;
        return p;
    };

    __bf16* sc_bf    = (__bf16*)alloc(BNC * 2);
    __bf16* mod1_wt  = (__bf16*)alloc(3145728 * 2);
    __bf16* mod2_wt  = (__bf16*)alloc(3145728 * 2);
    __bf16* qkv_ut   = (__bf16*)alloc(65536 * 2);
    __bf16* qkv_vc   = (__bf16*)alloc(3145728 * 2);
    __bf16* proj_ut  = (__bf16*)alloc(65536 * 2);
    __bf16* proj_vc  = (__bf16*)alloc(1048576 * 2);
    __bf16* fc1_wt   = (__bf16*)alloc(4194304 * 2);
    __bf16* fc2_wt   = (__bf16*)alloc(4194304 * 2);
    float*  m_f      = (float*)alloc(M3C * 4);
    float*  x1f      = (float*)alloc(BNC * 4);   // reused as x3f
    __bf16* x1h      = (__bf16*)alloc(BNC * 2);  // reused as x3h
    __bf16* t_bf     = (__bf16*)alloc(BNC * 2);
    __bf16* qkv_bf   = (__bf16*)alloc(QKV * 2);
    __bf16* o_bf     = (__bf16*)alloc(BNC * 2);
    __bf16* t2_bf    = (__bf16*)alloc(BNC * 2);
    float*  a_f      = (float*)alloc(BNC * 4);   // reused as MLP output y
    float*  xr_f     = (float*)alloc(BNC * 4);
    __bf16* h_bf     = (__bf16*)alloc(HEL * 2);

    // --- weight / activation conversions ---
    silu_bf16_kernel<<<4096, 256, 0, stream>>>(c, sc_bf, BNC);
    cvt_bf16_kernel<<<4096, 256, 0, stream>>>(qkv_v, qkv_vc, 3145728);
    cvt_bf16_kernel<<<2048, 256, 0, stream>>>(proj_v, proj_vc, 1048576);
    tconv_bf16_kernel<<<8192, 256, 0, stream>>>(mod1_w, mod1_wt, 3072, 1024);
    tconv_bf16_kernel<<<8192, 256, 0, stream>>>(mod2_w, mod2_wt, 3072, 1024);
    tconv_bf16_kernel<<<256, 256, 0, stream>>>(qkv_u, qkv_ut, 256, 256);
    tconv_bf16_kernel<<<256, 256, 0, stream>>>(proj_u, proj_ut, 256, 256);
    tconv_bf16_kernel<<<8192, 256, 0, stream>>>(fc1_w, fc1_wt, 4096, 1024);
    tconv_bf16_kernel<<<8192, 256, 0, stream>>>(fc2_w, fc2_wt, 1024, 4096);

    // --- G1: m1 = silu(c) @ mod1_w.T + mod1_b  (8192x3072x1024) ---
    gemm_bf16_kernel<false, true, false><<<dim3(12, 64, 1), 256, 0, stream>>>(
        sc_bf, mod1_wt, mod1_b, m_f, 8192, 3072, 1024, 0, 0, 0);
    // --- adaLN 1 ---
    adaln_kernel<<<8192, 256, 0, stream>>>(x, m_f, x1f, x1h);

    // --- attention: t = u^T @ x (batched 32x 256x1024x256) ---
    gemm_bf16_kernel<true, false, false><<<dim3(4, 2, 32), 256, 0, stream>>>(
        qkv_ut, x1h, nullptr, t_bf, 256, 1024, 256, 0, 262144, 262144);
    // --- qkv = t @ qkv_v (batched 32x 256x3072x1024) ---
    gemm_bf16_kernel<true, false, false><<<dim3(12, 2, 32), 256, 0, stream>>>(
        t_bf, qkv_vc, nullptr, qkv_bf, 256, 3072, 1024, 262144, 0, 786432);
    // --- per-head attention ---
    attention_kernel<<<32, 256, 0, stream>>>(qkv_bf, o_bf);
    // --- t2 = proj_u^T @ o (batched) ---
    gemm_bf16_kernel<true, false, false><<<dim3(4, 2, 32), 256, 0, stream>>>(
        proj_ut, o_bf, nullptr, t2_bf, 256, 1024, 256, 0, 262144, 262144);
    // --- a = t2 @ proj_v (batched 32x 256x1024x1024) ---
    gemm_bf16_kernel<false, false, false><<<dim3(4, 2, 32), 256, 0, stream>>>(
        t2_bf, proj_vc, nullptr, a_f, 256, 1024, 1024, 262144, 0, 262144);

    // --- x2 = x1 + gate1 * a ---
    residual_gate_kernel<<<8192, 256, 0, stream>>>(x1f, m_f, a_f, xr_f, BNC);

    // --- G4: m2 = silu(c) @ mod2_w.T + mod2_b ---
    gemm_bf16_kernel<false, true, false><<<dim3(12, 64, 1), 256, 0, stream>>>(
        sc_bf, mod2_wt, mod2_b, m_f, 8192, 3072, 1024, 0, 0, 0);
    // --- adaLN 2 (x3 into x1f/x1h) ---
    adaln_kernel<<<8192, 256, 0, stream>>>(xr_f, m_f, x1f, x1h);

    // --- MLP: h = gelu(x3 @ fc1_w.T + fc1_b)  (8192x4096x1024) ---
    gemm_bf16_kernel<true, true, true><<<dim3(16, 64, 1), 256, 0, stream>>>(
        x1h, fc1_wt, fc1_b, h_bf, 8192, 4096, 1024, 0, 0, 0);
    // --- y = h @ fc2_w.T + fc2_b  (8192x1024x4096) ---
    gemm_bf16_kernel<false, true, false><<<dim3(4, 64, 1), 256, 0, stream>>>(
        h_bf, fc2_wt, fc2_b, a_f, 8192, 1024, 4096, 0, 0, 0);

    // --- out = x3 + gate2 * y ---
    residual_gate_kernel<<<8192, 256, 0, stream>>>(x1f, m_f, a_f, out, BNC);
}
